// SuperposedExpert_26749056319684
// MI455X (gfx1250) — matrix operand order repaired and checked by reading, not compiled, not run on hardware
//
#include <hip/hip_runtime.h>
#include <hip/hip_bf16.h>
#include <math.h>
#include <cstdint>

typedef __bf16 bf16;
typedef __attribute__((ext_vector_type(16))) __bf16 v16bf;
typedef __attribute__((ext_vector_type(8)))  __bf16 v8bf;
typedef __attribute__((ext_vector_type(8)))  float  v8f;

#define B_   8192
#define D_   1024
#define F_   4096
#define S_   4
#define H_   4
#define HD_  256

// ---------------- prep kernels ----------------
__global__ __launch_bounds__(256) void f32_to_bf16_kernel(const float* __restrict__ src,
                                                          bf16* __restrict__ dst, long n) {
  long i = (long)blockIdx.x * 256 + threadIdx.x;
  if (i < n) dst[i] = (bf16)src[i];
}

// dst[n*rows + k] = (bf16) src[k*cols + n]   (store N-major for WMMA B operand)
__global__ __launch_bounds__(256) void transpose_to_bf16_kernel(const float* __restrict__ src,
                                                                bf16* __restrict__ dst,
                                                                int rows /*K*/, int cols /*N*/) {
  long idx = (long)blockIdx.x * 256 + threadIdx.x;
  if (idx >= (long)rows * cols) return;
  int k = (int)(idx % rows);
  int n = (int)(idx / rows);
  dst[idx] = (bf16)src[(long)k * cols + n];
}

// w1T[s, o1*64+o2, i1*32+i2] = sum_r g1a[s,i1,o1,r] * g1b[s,r,i2,o2]   (N-major)
__global__ __launch_bounds__(256) void build_w1T_kernel(const float* __restrict__ g1a,
                                                        const float* __restrict__ g1b,
                                                        bf16* __restrict__ w1T) {
  long idx = (long)blockIdx.x * 256 + threadIdx.x;
  if (idx >= (long)S_ * F_ * D_) return;
  int i  = (int)(idx % D_);          // K index (d_model)
  long t = idx / D_;
  int o  = (int)(t % F_);            // N index (d_ff)
  int s  = (int)(t / F_);
  int i1 = i >> 5, i2 = i & 31;
  int o1 = o >> 6, o2 = o & 63;
  const float* a = g1a + (((long)s * 32 + i1) * 64 + o1) * 16;          // +r
  const float* b = g1b + ((long)s * 16) * 2048 + (long)i2 * 64 + o2;    // +r*2048
  float acc = 0.f;
#pragma unroll
  for (int r = 0; r < 16; ++r) acc += a[r] * b[(long)r * 2048];
  w1T[idx] = (bf16)acc;
}

// w2T[s, i1*32+i2, o1*64+o2] = sum_r g2a[s,o1,i1,r] * g2b[s,r,o2,i2]   (N-major: N=d_model, K=d_ff)
__global__ __launch_bounds__(256) void build_w2T_kernel(const float* __restrict__ g2a,
                                                        const float* __restrict__ g2b,
                                                        bf16* __restrict__ w2T) {
  long idx = (long)blockIdx.x * 256 + threadIdx.x;
  if (idx >= (long)S_ * D_ * F_) return;
  int f  = (int)(idx % F_);          // K index (d_ff)
  long t = idx / F_;
  int d  = (int)(t % D_);            // N index (d_model)
  int s  = (int)(t / D_);
  int o1 = f >> 6, o2 = f & 63;
  int i1 = d >> 5, i2 = d & 31;
  const float* a = g2a + (((long)s * 64 + o1) * 32 + i1) * 16;          // +r
  const float* b = g2b + ((long)s * 16) * 2048 + (long)o2 * 32 + i2;    // +r*2048
  float acc = 0.f;
#pragma unroll
  for (int r = 0; r < 16; ++r) acc += a[r] * b[(long)r * 2048];
  w2T[idx] = (bf16)acc;
}

// ---------------- WMMA GEMM (async double-buffered) ----------------
__device__ __forceinline__ float gelu_exact(float v) {
  return 0.5f * v * (1.0f + erff(v * 0.70710678118654752f));
}

__device__ __forceinline__ unsigned lds_off(const void* p) {
  return (unsigned)(uintptr_t)p;   // low 32 bits of flat LDS address = LDS byte offset
}

__device__ __forceinline__ void async_b128(unsigned lds_addr, const void* gaddr) {
  asm volatile("global_load_async_to_lds_b128 %0, %1, off"
               :: "v"(lds_addr), "v"((unsigned long long)(uintptr_t)gaddr)
               : "memory");
}

#define BM 128
#define BN 128
#define BK 64
#define LDA_S 72   // LDS row stride in bf16 (144B, 16B-aligned), padded vs 64
#define LDB_S 72

// A: [M][K] row-major (K-contiguous).  BT: [N][K] row-major (pre-transposed weights).
// EPI: 0 = none, 1 = exact gelu, 2 = +bias[n]
template <int EPI, typename OutT>
__global__ __launch_bounds__(256) void gemm_bf16_wmma(
    const bf16* __restrict__ A, const bf16* __restrict__ BT,
    const float* __restrict__ bias, OutT* __restrict__ C,
    int M, int N, int K, long sA, long sB, long sC) {
  __shared__ bf16 As[2][BM * LDA_S];
  __shared__ bf16 Bs[2][BN * LDB_S];

  const int tid  = threadIdx.x;
  const int lane = tid & 31;
  const int wave = tid >> 5;        // 8 waves: 2 x 4 wave tiles
  const int wr   = wave >> 2;       // 64 rows each
  const int wc   = wave & 3;        // 32 cols each
  const int bm0  = blockIdx.x * BM;
  const int bn0  = blockIdx.y * BN;
  const int hh   = lane >> 4;       // lane half
  const int l16  = lane & 15;

  A  += (long)blockIdx.z * sA;
  BT += (long)blockIdx.z * sB;
  C  += (long)blockIdx.z * sC;

  v8f acc[4][2];
#pragma unroll
  for (int i = 0; i < 4; ++i)
#pragma unroll
    for (int j = 0; j < 2; ++j) acc[i][j] = (v8f){0.f,0.f,0.f,0.f,0.f,0.f,0.f,0.f};

  // issue one double-buffer stage: 128x64 bf16 tiles of A and BT, 16B per lane per op,
  // 4+4 async ops per thread on ASYNCcnt
  auto issue_stage = [&](int st, int k0) {
#pragma unroll
    for (int it = 0; it < 4; ++it) {
      int c = tid + it * 256;            // 0..1023 chunks (128 rows x 8 chunks)
      int row = c >> 3;
      int kk  = (c & 7) << 3;
      async_b128(lds_off(&As[st][row * LDA_S + kk]),
                 A + (long)(bm0 + row) * K + k0 + kk);
    }
#pragma unroll
    for (int it = 0; it < 4; ++it) {
      int c = tid + it * 256;
      int n  = c >> 3;
      int kk = (c & 7) << 3;
      async_b128(lds_off(&Bs[st][n * LDB_S + kk]),
                 BT + (long)(bn0 + n) * K + k0 + kk);
    }
  };

  // consume one stage: 2 WMMA K-slices; per slice load ALL 6 fragments (12x ds_load_b128)
  // into distinct registers first, then drain 8 WMMAs (avoids WAR-forced dscnt waits)
  auto compute_stage = [&](int cur) {
#pragma unroll
    for (int t = 0; t < 2; ++t) {
      v16bf bfrag[2];
      v16bf afrag[4];
#pragma unroll
      for (int j = 0; j < 2; ++j) {
        // B fragment: lane n = l16; elem e -> K = 32t + 16*hh + e (contiguous)
        const bf16* bp = &Bs[cur][(wc * 32 + j * 16 + l16) * LDB_S + 32 * t + 16 * hh];
        v8bf lo = *(const v8bf*)bp;
        v8bf hi = *(const v8bf*)(bp + 8);
        bfrag[j] = __builtin_shufflevector(lo, hi, 0,1,2,3,4,5,6,7,8,9,10,11,12,13,14,15);
      }
#pragma unroll
      for (int i = 0; i < 4; ++i) {
        // A fragment: lane m = l16; elem e -> K = 32t + 16*hh + e (contiguous)
        const bf16* ap = &As[cur][(wr * 64 + i * 16 + l16) * LDA_S + 32 * t];
        v8bf lo = *(const v8bf*)(ap + 16 * hh);
        v8bf hi = *(const v8bf*)(ap + 16 * hh + 8);
        afrag[i] = __builtin_shufflevector(lo, hi, 0,1,2,3,4,5,6,7,8,9,10,11,12,13,14,15);
      }
#pragma unroll
      for (int i = 0; i < 4; ++i)
#pragma unroll
        for (int j = 0; j < 2; ++j)
          acc[i][j] = __builtin_amdgcn_wmma_f32_16x16x32_bf16(
              false, afrag[i], false, bfrag[j], (short)0, acc[i][j], false, false);
    }
  };

  const int nsteps = K / BK;
  issue_stage(0, 0);

  // steady state: branch-free issue -> wait(8) -> barrier -> compute -> barrier
  for (int step = 0; step < nsteps - 1; ++step) {
    const int cur = step & 1;
    issue_stage(cur ^ 1, (step + 1) * BK);
    asm volatile("s_wait_asynccnt 8" ::: "memory");   // this wave's stage `cur` landed
    __syncthreads();                                  // whole WG's stage `cur` landed
    compute_stage(cur);
    __syncthreads();                                  // done reading before refill
  }
  // peeled last stage
  asm volatile("s_wait_asynccnt 0" ::: "memory");
  __syncthreads();
  compute_stage((nsteps - 1) & 1);

  // epilogue + store: D layout m = vgpr + 8*half, n = lane%16
#pragma unroll
  for (int i = 0; i < 4; ++i) {
#pragma unroll
    for (int j = 0; j < 2; ++j) {
      int n = bn0 + wc * 32 + j * 16 + l16;
      float bv = (EPI == 2) ? bias[n] : 0.f;
#pragma unroll
      for (int v = 0; v < 8; ++v) {
        int m = bm0 + wr * 64 + i * 16 + hh * 8 + v;
        float val = acc[i][j][v];
        if (EPI == 2) val += bv;
        if (EPI == 1) val = gelu_exact(val);
        C[(long)m * N + n] = (OutT)val;
      }
    }
  }
}

// ---------------- softmax collapse over S ----------------
__global__ __launch_bounds__(256) void attn_collapse_kernel(
    const float* __restrict__ q, const bf16* __restrict__ kbuf,
    const bf16* __restrict__ vbuf, bf16* __restrict__ obuf) {
  int gw   = (int)((blockIdx.x * 256 + threadIdx.x) >> 5);
  int lane = threadIdx.x & 31;
  int b = gw / H_, h = gw % H_;
  const float* qv = q + (long)b * D_ + h * HD_;
  const float scale = 0.0625f;  // 1/sqrt(256)
  float sc[S_];
#pragma unroll
  for (int s = 0; s < S_; ++s) {
    const bf16* kv = kbuf + ((long)s * B_ + b) * D_ + h * HD_;
    float p = 0.f;
#pragma unroll
    for (int e = lane; e < HD_; e += 32) p += qv[e] * (float)kv[e];
#pragma unroll
    for (int off = 16; off > 0; off >>= 1) p += __shfl_xor(p, off, 32);
    sc[s] = p * scale;
  }
  float mx = fmaxf(fmaxf(sc[0], sc[1]), fmaxf(sc[2], sc[3]));
  float w[S_], sum = 0.f;
#pragma unroll
  for (int s = 0; s < S_; ++s) { w[s] = expf(sc[s] - mx); sum += w[s]; }
  float inv = 1.f / sum;
  bf16* ov = obuf + (long)b * D_ + h * HD_;
#pragma unroll
  for (int e = lane; e < HD_; e += 32) {
    float acc = 0.f;
#pragma unroll
    for (int s = 0; s < S_; ++s)
      acc += (w[s] * inv) * (float)vbuf[((long)s * B_ + b) * D_ + h * HD_ + e];
    ov[e] = (bf16)acc;
  }
}

// ---------------- launch ----------------
extern "C" void kernel_launch(void* const* d_in, const int* in_sizes, int n_in,
                              void* d_out, int out_size, void* d_ws, size_t ws_size,
                              hipStream_t stream) {
  (void)in_sizes; (void)n_in; (void)out_size; (void)ws_size;
  const float* x   = (const float*)d_in[0];
  const float* g1a = (const float*)d_in[1];
  const float* g1b = (const float*)d_in[2];
  const float* g2a = (const float*)d_in[3];
  const float* g2b = (const float*)d_in[4];
  const float* wq  = (const float*)d_in[5];
  const float* bq  = (const float*)d_in[6];
  const float* wk  = (const float*)d_in[7];
  const float* bk  = (const float*)d_in[8];
  const float* wv  = (const float*)d_in[9];
  const float* bv  = (const float*)d_in[10];
  const float* wo  = (const float*)d_in[11];
  const float* bo  = (const float*)d_in[12];
  float* out = (float*)d_out;

  char* ws = (char*)d_ws;
  size_t off = 0;
  auto alloc = [&](size_t bytes) -> void* {
    void* p = ws + off;
    off += (bytes + 255) & ~(size_t)255;
    return p;
  };
  bf16* x_bf   = (bf16*)alloc((size_t)B_ * D_ * 2);
  bf16* wqT_bf = (bf16*)alloc((size_t)D_ * D_ * 2);
  bf16* wkT_bf = (bf16*)alloc((size_t)D_ * D_ * 2);
  bf16* wvT_bf = (bf16*)alloc((size_t)D_ * D_ * 2);
  bf16* woT_bf = (bf16*)alloc((size_t)D_ * D_ * 2);
  bf16* w1T_bf = (bf16*)alloc((size_t)S_ * F_ * D_ * 2);
  bf16* w2T_bf = (bf16*)alloc((size_t)S_ * D_ * F_ * 2);
  bf16* h_bf   = (bf16*)alloc((size_t)S_ * B_ * F_ * 2);
  bf16* y_bf   = (bf16*)alloc((size_t)S_ * B_ * D_ * 2);
  float* q_f   = (float*)alloc((size_t)B_ * D_ * 4);
  bf16* k_bf   = (bf16*)alloc((size_t)S_ * B_ * D_ * 2);
  bf16* v_bf   = (bf16*)alloc((size_t)S_ * B_ * D_ * 2);
  bf16* o_bf   = (bf16*)alloc((size_t)B_ * D_ * 2);

  // converts: x row-major (A side); projections transposed to N-major (B side)
  {
    long nx = (long)B_ * D_;
    f32_to_bf16_kernel<<<(nx + 255) / 256, 256, 0, stream>>>(x, x_bf, nx);
    long nw = (long)D_ * D_;
    transpose_to_bf16_kernel<<<(nw + 255) / 256, 256, 0, stream>>>(wq, wqT_bf, D_, D_);
    transpose_to_bf16_kernel<<<(nw + 255) / 256, 256, 0, stream>>>(wk, wkT_bf, D_, D_);
    transpose_to_bf16_kernel<<<(nw + 255) / 256, 256, 0, stream>>>(wv, wvT_bf, D_, D_);
    transpose_to_bf16_kernel<<<(nw + 255) / 256, 256, 0, stream>>>(wo, woT_bf, D_, D_);
  }
  // TT weight materialization (directly N-major)
  {
    long n1 = (long)S_ * D_ * F_;
    build_w1T_kernel<<<(n1 + 255) / 256, 256, 0, stream>>>(g1a, g1b, w1T_bf);
    build_w2T_kernel<<<(n1 + 255) / 256, 256, 0, stream>>>(g2a, g2b, w2T_bf);
  }
  // h = gelu(x @ w1[s])   M=8192, N=4096, K=1024
  gemm_bf16_wmma<1, bf16><<<dim3(B_ / BM, F_ / BN, S_), 256, 0, stream>>>(
      x_bf, w1T_bf, nullptr, h_bf, B_, F_, D_,
      0L, (long)F_ * D_, (long)B_ * F_);
  // y = h @ w2[s]          M=8192, N=1024, K=4096
  gemm_bf16_wmma<0, bf16><<<dim3(B_ / BM, D_ / BN, S_), 256, 0, stream>>>(
      h_bf, w2T_bf, nullptr, y_bf, B_, D_, F_,
      (long)B_ * F_, (long)D_ * F_, (long)B_ * D_);
  // q = x @ wq + bq  (f32 for attention)
  gemm_bf16_wmma<2, float><<<dim3(B_ / BM, D_ / BN, 1), 256, 0, stream>>>(
      x_bf, wqT_bf, bq, q_f, B_, D_, D_, 0L, 0L, 0L);
  // k = y @ wk + bk ; v = y @ wv + bv  (batched over s)
  gemm_bf16_wmma<2, bf16><<<dim3(B_ / BM, D_ / BN, S_), 256, 0, stream>>>(
      y_bf, wkT_bf, bk, k_bf, B_, D_, D_, (long)B_ * D_, 0L, (long)B_ * D_);
  gemm_bf16_wmma<2, bf16><<<dim3(B_ / BM, D_ / BN, S_), 256, 0, stream>>>(
      y_bf, wvT_bf, bv, v_bf, B_, D_, D_, (long)B_ * D_, 0L, (long)B_ * D_);
  // softmax over S and collapse: one wave per (b, head)
  attn_collapse_kernel<<<dim3((B_ * H_) / 8), 256, 0, stream>>>(q_f, k_bf, v_bf, o_bf);
  // out = o @ wo + bo (f32 to d_out)
  gemm_bf16_wmma<2, float><<<dim3(B_ / BM, D_ / BN, 1), 256, 0, stream>>>(
      o_bf, woT_bf, bo, out, B_, D_, D_, 0L, 0L, 0L);
}
